// EncoderBlock_74311524156045
// MI455X (gfx1250) — matrix operand order repaired and checked by reading, not compiled, not run on hardware
//
#include <hip/hip_runtime.h>
#include <hip/hip_bf16.h>
#include <math.h>

// ---- problem constants (match reference) ----
constexpr int kB    = 4;
constexpr int kS    = 2048;
constexpr int kEmb  = 1024;
constexpr int kHid  = 64;
constexpr int kHead = 16;
constexpr int kFfw  = 4096;
constexpr int kBS   = kB * kS;          // 8192 rows
constexpr float kEps = 1e-5f;

typedef __bf16 bf16_t;
typedef __attribute__((ext_vector_type(16))) __bf16 v16bf;
typedef __attribute__((ext_vector_type(8)))  float  v8f;

// ---- CDNA5 async global->LDS copy (guarded; falls back to sync copy) ----
#if defined(__has_builtin)
#  if __has_builtin(__builtin_amdgcn_global_load_async_to_lds_b128) && \
      __has_builtin(__builtin_amdgcn_s_wait_asynccnt)
#    define ATHENA_ASYNC 1
#  endif
#endif
#ifndef ATHENA_ASYNC
#  define ATHENA_ASYNC 0
#endif

#if ATHENA_ASYNC
typedef int a_v4i __attribute__((vector_size(16)));
typedef __attribute__((address_space(1))) a_v4i* g_v4i_p;   // global (AS1)
typedef __attribute__((address_space(3))) a_v4i* l_v4i_p;   // LDS (AS3)
__device__ __forceinline__ void g2l_b128(const bf16_t* g, bf16_t* l) {
    // direct memory->LDS, tracked by ASYNCcnt, no VGPR round trip
    __builtin_amdgcn_global_load_async_to_lds_b128((g_v4i_p)g, (l_v4i_p)l, 0, 0);
}
__device__ __forceinline__ void async_wait0() { __builtin_amdgcn_s_wait_asynccnt(0); }
#else
__device__ __forceinline__ void g2l_b128(const bf16_t* g, bf16_t* l) {
    *(uint4*)l = *(const uint4*)g;     // 16B through VGPRs (compiler inserts waits)
}
__device__ __forceinline__ void async_wait0() {}
#endif

// ---- helpers ----
__device__ __forceinline__ bf16_t f2bf(float f) {
    unsigned u = __float_as_uint(f);
    u += 0x7FFFu + ((u >> 16) & 1u);            // round-to-nearest-even
    unsigned short h = (unsigned short)(u >> 16);
    bf16_t r;
    __builtin_memcpy(&r, &h, 2);
    return r;
}

__device__ __forceinline__ v8f wmma_bf16(v16bf a, v16bf b, v8f c) {
    // D = A(16x32 bf16) * B(32x16 bf16) + C(16x16 f32)
    return __builtin_amdgcn_wmma_f32_16x16x32_bf16(
        /*neg_a=*/false, a, /*neg_b=*/false, b,
        /*c_mod=*/(short)0, c, /*reuse_a=*/false, /*reuse_b=*/false);
}

// A-fragment (16x32, bf16): lane%16 = row M, lane/16 = K-half.
// VGPR i, sub s -> K = half*16 + i*2 + s   (pairs contiguous)
__device__ __forceinline__ v16bf load_a_frag(const bf16_t* base, int stride, int r, int hf) {
    union { v16bf v; bf16_t e[16]; } u;
    const bf16_t* row = base + r * stride;
#pragma unroll
    for (int i = 0; i < 8; ++i) {
        int k = hf * 16 + i * 2;
        u.e[2 * i]     = row[k];
        u.e[2 * i + 1] = row[k + 1];
    }
    return u.v;
}

// B-fragment (32x16, bf16) from a [N][K]-transposed tile (no swizzle):
// lane%16 = col N, lane/16 = K-half.  VGPR i, sub s -> K = half*16 + i*2 + s
__device__ __forceinline__ v16bf load_b_fragT(const bf16_t* base, int stride, int r, int hf) {
    union { v16bf v; bf16_t e[16]; } u;
    const bf16_t* row = base + r * stride;
#pragma unroll
    for (int i = 0; i < 8; ++i) {
        int k = hf * 16 + i * 2;
        u.e[2 * i]     = row[k];
        u.e[2 * i + 1] = row[k + 1];
    }
    return u.v;
}

// Same, but the tile uses an XOR chunk swizzle: logical 8-elem chunk c of row n
// lives at physical chunk c ^ sw (sw = (tile_base>>4)&3, constant per 16-row tile).
__device__ __forceinline__ v16bf load_b_frag_sw(const bf16_t* base, int stride, int r,
                                                int hf, int sw) {
    union { v16bf v; bf16_t e[16]; } u;
    const bf16_t* row = base + r * stride;
#pragma unroll
    for (int i = 0; i < 8; ++i) {
        int k = hf * 16 + i * 2;
        int phys = ((((k >> 3) & 3) ^ sw) << 3) | (k & 7);   // pairs stay inside a chunk
        u.e[2 * i]     = row[phys];
        u.e[2 * i + 1] = row[phys + 1];
    }
    return u.v;
}

// butterfly reductions within 16-lane half-groups (xor masks 1..8 stay in-group)
__device__ __forceinline__ float rmax16(float v) {
#pragma unroll
    for (int m = 8; m >= 1; m >>= 1) v = fmaxf(v, __shfl_xor(v, m, 32));
    return v;
}
__device__ __forceinline__ float rsum16(float v) {
#pragma unroll
    for (int m = 8; m >= 1; m >>= 1) v += __shfl_xor(v, m, 32);
    return v;
}

// ---- f32 -> bf16 conversion ----
__global__ __launch_bounds__(256) void cvt_bf16_kernel(const float* __restrict__ in,
                                                       bf16_t* __restrict__ out, int n) {
    int i = blockIdx.x * blockDim.x + threadIdx.x;
    int stride = gridDim.x * blockDim.x;
    for (; i < n; i += stride) out[i] = f2bf(in[i]);
}

// ---- generic bf16 WMMA GEMM: out = act((A @ B + bias)) * alpha ----
// 128x128 block tile, 8 waves (2x4), wave tile 64x32 (4x2 C tiles), K staged
// 32-wide, double-buffered. A: async global->LDS b128. B: coalesced b128 global
// reads + bank-swizzled transposed b16 LDS stores (frag reads stay ds_load_b128).
__global__ __launch_bounds__(256) void gemm_bf16_kernel(
    const bf16_t* __restrict__ A, const bf16_t* __restrict__ Bm,
    const float* __restrict__ bias, float* __restrict__ outF,
    bf16_t* __restrict__ outB, int M, int N, int K, float alpha, int act) {
    __shared__ bf16_t Alds[2][128 * 40];   // [row][k], 80B row stride (16B-aligned)
    __shared__ bf16_t Blds[2][128 * 40];   // transposed [n][k], chunk-swizzled

    const int tid  = threadIdx.x;
    const int lane = tid & 31, wave = tid >> 5;
    const int r = lane & 15, hf = lane >> 4;
    const int wm = wave >> 2, wn = wave & 3;     // 2x4 wave grid
    const int m0 = blockIdx.y * 128, n0 = blockIdx.x * 128;

    v8f acc[4][2];
#pragma unroll
    for (int mt = 0; mt < 4; ++mt)
#pragma unroll
        for (int nt = 0; nt < 2; ++nt) acc[mt][nt] = {};

    const int ar  = tid >> 1, ac0 = (tid & 1) * 16;   // A: 2 threads per 32-wide row
    const int bkk = tid >> 3, bn0 = (tid & 7) * 16;   // B: 8 threads per 128-wide row

    auto stage = [&](int k0, int buf) {
        // A tile: two async 16B chunks per thread
        const bf16_t* Ag = A + (size_t)(m0 + ar) * K + k0 + ac0;
        g2l_b128(Ag,     &Alds[buf][ar * 40 + ac0]);
        g2l_b128(Ag + 8, &Alds[buf][ar * 40 + ac0 + 8]);
        // B tile: coalesced b128 reads, swizzled transposed b16 stores
        const bf16_t* Bg = Bm + (size_t)(k0 + bkk) * N + n0 + bn0;
        bf16_t tmp[16];
        *(uint4*)&tmp[0] = *(const uint4*)Bg;
        *(uint4*)&tmp[8] = *(const uint4*)(Bg + 8);
#pragma unroll
        for (int t = 0; t < 16; ++t) {
            int n  = bn0 + t;
            int pc = ((((bkk >> 3) & 3) ^ ((n >> 4) & 3)) << 3) | (bkk & 7);
            Blds[buf][n * 40 + pc] = tmp[t];
        }
        if (k0 + 64 < K) {
            __builtin_prefetch(A + (size_t)(m0 + ar) * K + k0 + 64 + ac0, 0, 1);
            __builtin_prefetch(Bm + (size_t)(k0 + 64 + bkk) * N + n0 + bn0, 0, 1);
        }
    };
    auto compute = [&](int buf) {
        v16bf af[4], bfg[2];
#pragma unroll
        for (int mt = 0; mt < 4; ++mt)
            af[mt] = load_a_frag(&Alds[buf][(wm * 64 + mt * 16) * 40], 40, r, hf);
#pragma unroll
        for (int nt = 0; nt < 2; ++nt)
            bfg[nt] = load_b_frag_sw(&Blds[buf][(wn * 32 + nt * 16) * 40], 40, r, hf,
                                     (wn * 2 + nt) & 3);
#pragma unroll
        for (int mt = 0; mt < 4; ++mt)
#pragma unroll
            for (int nt = 0; nt < 2; ++nt)
                acc[mt][nt] = wmma_bf16(af[mt], bfg[nt], acc[mt][nt]);
    };

    const int nk = K / 32;                        // 32 or 128: always even
    stage(0, 0);
    for (int it = 0; it < nk; it += 2) {
        async_wait0();                            // buf0's asyncs done
        __syncthreads();                          // buf1 free (read 1 iter ago)
        if (it + 1 < nk) stage((it + 1) * 32, 1); // overlap copy with compute
        compute(0);
        async_wait0();
        __syncthreads();
        if (it + 2 < nk) stage((it + 2) * 32, 0);
        compute(1);
    }

    // epilogue: C layout row = half*8 + i, col = lane%16
#pragma unroll
    for (int mt = 0; mt < 4; ++mt)
#pragma unroll
        for (int nt = 0; nt < 2; ++nt) {
            int row0 = m0 + wm * 64 + mt * 16 + hf * 8;
            int col  = n0 + wn * 32 + nt * 16 + r;
            float bia = bias ? bias[col] : 0.f;
#pragma unroll
            for (int i = 0; i < 8; ++i) {
                float v = acc[mt][nt][i] + bia;
                if (act == 1) v = 0.5f * v * (1.f + erff(v * 0.70710678118654752f)); // exact GELU
                v *= alpha;
                size_t idx = (size_t)(row0 + i) * N + col;
                if (outF) outF[idx] = v;
                if (outB) outB[idx] = f2bf(v);
            }
        }
}

// ---- fused flash attention ----
// q/k/v: [B*S, H*D] bf16 (col = h*64+f). q pre-scaled by 1/sqrt(D).
// block = (b, h, 128 q-rows); 8 waves x 16 q-rows; 32 keys/step, double-buffered.
__global__ __launch_bounds__(256) void attn_kernel(
    const bf16_t* __restrict__ qb, const bf16_t* __restrict__ kb,
    const bf16_t* __restrict__ vb, const unsigned char* __restrict__ mask,
    bf16_t* __restrict__ zb) {
    __shared__ bf16_t Kt[2][32 * 72];     // [key][feat], 144B rows (16B-aligned)
    __shared__ bf16_t Vt[2][64 * 40];     // transposed [feat][key], chunk-swizzled
    __shared__ bf16_t Pl[8 * 16 * 40];    // per-wave P tile [qrow][key] (wave-private)

    const int tid = threadIdx.x, lane = tid & 31, wave = tid >> 5;
    const int r = lane & 15, hf = lane >> 4;
    const int qt = blockIdx.x & 15;             // S/128 = 16 q-tiles
    const int bh = blockIdx.x >> 4;
    const int h = bh & 15, b = bh >> 4;
    const int qrow0 = qt * 128 + wave * 16;

    // Q A-fragments for this wave's 16 rows (two K=32 chunks over D=64)
    v16bf aq[2];
    {
        const bf16_t* qrow = qb + (size_t)(b * kS + qrow0 + r) * kEmb + h * kHid;
#pragma unroll
        for (int kh = 0; kh < 2; ++kh) {
            union { v16bf v; bf16_t e[16]; } u;
#pragma unroll
            for (int i = 0; i < 8; ++i) {
                int f = kh * 32 + hf * 16 + i * 2;
                u.e[2 * i] = qrow[f];
                u.e[2 * i + 1] = qrow[f + 1];
            }
            aq[kh] = u.v;
        }
    }

    v8f oacc[4];
#pragma unroll
    for (int ft = 0; ft < 4; ++ft) oacc[ft] = {};
    float mrun[8], lrun[8];
#pragma unroll
    for (int i = 0; i < 8; ++i) { mrun[i] = -1e30f; lrun[i] = 0.f; }

    const int skey = tid >> 3, sf0 = (tid & 7) * 8;   // staging: 8 feats/thread

    auto stage_kv = [&](int j, int buf) {
        const bf16_t* krow = kb + (size_t)(b * kS + j * 32 + skey) * kEmb + h * kHid + sf0;
        const bf16_t* vrow = vb + (size_t)(b * kS + j * 32 + skey) * kEmb + h * kHid + sf0;
        // K tile: straight copy -> async b128 directly into LDS
        g2l_b128(krow, &Kt[buf][skey * 72 + sf0]);
        // V tile: transpose through VGPRs, swizzled stores
        bf16_t vt[8];
        *(uint4*)vt = *(const uint4*)vrow;
#pragma unroll
        for (int t = 0; t < 8; ++t) {
            int n  = sf0 + t;                     // feature row
            int pc = ((((skey >> 3) & 3) ^ ((n >> 4) & 3)) << 3) | (skey & 7);
            Vt[buf][n * 40 + pc] = vt[t];
        }
    };

    auto process = [&](int j, int buf) {
        // scores: two 16x16 tiles (32 keys), K-dim = D = 64 -> 2 WMMAs each
        v8f sc[2];
        sc[0] = {}; sc[1] = {};
#pragma unroll
        for (int sub = 0; sub < 2; ++sub)
#pragma unroll
            for (int kh = 0; kh < 2; ++kh) {
                v16bf bkf = load_b_fragT(&Kt[buf][(sub * 16) * 72 + kh * 32], 72, r, hf);
                sc[sub] = wmma_bf16(aq[kh], bkf, sc[sub]);
            }

        // mask + online softmax (row stats per C-layout row = hf*8+i)
#pragma unroll
        for (int i = 0; i < 8; ++i) {
            int qr = qt * 128 + wave * 16 + hf * 8 + i;
            size_t mbase = (size_t)b * kS * kS + (size_t)qr * kS;
            float s0 = sc[0][i], s1 = sc[1][i];
            s0 = mask[mbase + j * 32 + r]      ? -1e30f : s0;
            s1 = mask[mbase + j * 32 + 16 + r] ? -1e30f : s1;
            float mc  = rmax16(fmaxf(s0, s1));
            float mn  = fmaxf(mrun[i], mc);
            float scf = __expf(mrun[i] - mn);
            float p0 = __expf(s0 - mn), p1 = __expf(s1 - mn);
            lrun[i] = lrun[i] * scf + rsum16(p0 + p1);
            mrun[i] = mn;
#pragma unroll
            for (int ft = 0; ft < 4; ++ft) oacc[ft][i] *= scf;
            Pl[(wave * 16 + hf * 8 + i) * 40 + r]      = f2bf(p0);
            Pl[(wave * 16 + hf * 8 + i) * 40 + 16 + r] = f2bf(p1);
        }

        // P (16x32) @ V (32x64): redistribute P via per-wave LDS, 4 WMMAs
        v16bf ap = load_a_frag(&Pl[wave * 16 * 40], 40, r, hf);
#pragma unroll
        for (int ft = 0; ft < 4; ++ft) {
            v16bf bvf = load_b_frag_sw(&Vt[buf][(ft * 16) * 40], 40, r, hf, ft & 3);
            oacc[ft] = wmma_bf16(ap, bvf, oacc[ft]);
        }
    };

    const int nj = kS / 32;                       // 64: even
    stage_kv(0, 0);
    for (int j = 0; j < nj; j += 2) {
        async_wait0();
        __syncthreads();
        if (j + 1 < nj) stage_kv(j + 1, 1);       // overlap copy with compute
        process(j, 0);
        async_wait0();
        __syncthreads();
        if (j + 2 < nj) stage_kv(j + 2, 0);
        process(j + 1, 1);
    }

    // finalize: z[b,s,h*64+f] (== transpose(0,2,1,3).reshape of reference)
#pragma unroll
    for (int i = 0; i < 8; ++i) {
        float inv = lrun[i] > 0.f ? 1.f / lrun[i] : 0.f;
        int row = b * kS + qt * 128 + wave * 16 + hf * 8 + i;
#pragma unroll
        for (int ft = 0; ft < 4; ++ft)
            zb[(size_t)row * kEmb + h * kHid + ft * 16 + r] = f2bf(oacc[ft][i] * inv);
    }
}

// ---- residual add + layernorm (one row per block) ----
__global__ __launch_bounds__(256) void add_ln_kernel(
    const float* __restrict__ x, const float* __restrict__ y,
    const float* __restrict__ g, const float* __restrict__ be,
    float* __restrict__ outF, bf16_t* __restrict__ outB) {
    const int row = blockIdx.x, tid = threadIdx.x;
    __shared__ float red[256];
    const float* xr = x + (size_t)row * kEmb;
    const float* yr = y + (size_t)row * kEmb;
    float v[4];
    float s = 0.f;
#pragma unroll
    for (int c = 0; c < 4; ++c) {
        int col = tid + 256 * c;
        v[c] = xr[col] + yr[col];
        s += v[c];
    }
    red[tid] = s;
    __syncthreads();
    for (int st = 128; st > 0; st >>= 1) {
        if (tid < st) red[tid] += red[tid + st];
        __syncthreads();
    }
    float mu = red[0] * (1.f / kEmb);
    __syncthreads();
    float sq = 0.f;
#pragma unroll
    for (int c = 0; c < 4; ++c) { float d = v[c] - mu; sq += d * d; }
    red[tid] = sq;
    __syncthreads();
    for (int st = 128; st > 0; st >>= 1) {
        if (tid < st) red[tid] += red[tid + st];
        __syncthreads();
    }
    float rsig = rsqrtf(red[0] * (1.f / kEmb) + kEps);
#pragma unroll
    for (int c = 0; c < 4; ++c) {
        int col = tid + 256 * c;
        float o = (v[c] - mu) * rsig * g[col] + be[col];
        size_t idx = (size_t)row * kEmb + col;
        if (outF) outF[idx] = o;
        if (outB) outB[idx] = f2bf(o);
    }
}

// ---- host orchestration ----
extern "C" void kernel_launch(void* const* d_in, const int* in_sizes, int n_in,
                              void* d_out, int out_size, void* d_ws, size_t ws_size,
                              hipStream_t stream) {
    (void)in_sizes; (void)n_in; (void)out_size; (void)ws_size;
    const float* x  = (const float*)d_in[0];
    const unsigned char* mask = (const unsigned char*)d_in[1];
    const float* Wq = (const float*)d_in[2];  const float* bq = (const float*)d_in[3];
    const float* Wk = (const float*)d_in[4];  const float* bk = (const float*)d_in[5];
    const float* Wv = (const float*)d_in[6];  const float* bv = (const float*)d_in[7];
    const float* Wo = (const float*)d_in[8];  const float* bo = (const float*)d_in[9];
    const float* W1 = (const float*)d_in[10]; const float* b1 = (const float*)d_in[11];
    const float* W2 = (const float*)d_in[12]; const float* b2 = (const float*)d_in[13];
    const float* g1 = (const float*)d_in[14]; const float* be1 = (const float*)d_in[15];
    const float* g2 = (const float*)d_in[16]; const float* be2 = (const float*)d_in[17];

    char* ws = (char*)d_ws;
    size_t off = 0;
    auto alloc = [&](size_t bytes) {
        size_t o = off;
        off += (bytes + 255) & ~(size_t)255;
        return ws + o;
    };

    bf16_t* xb  = (bf16_t*)alloc((size_t)kBS * kEmb * 2);
    bf16_t* Wqb = (bf16_t*)alloc((size_t)kEmb * kEmb * 2);
    bf16_t* Wkb = (bf16_t*)alloc((size_t)kEmb * kEmb * 2);
    bf16_t* Wvb = (bf16_t*)alloc((size_t)kEmb * kEmb * 2);
    bf16_t* Wob = (bf16_t*)alloc((size_t)kEmb * kEmb * 2);
    bf16_t* W1b = (bf16_t*)alloc((size_t)kEmb * kFfw * 2);
    bf16_t* W2b = (bf16_t*)alloc((size_t)kFfw * kEmb * 2);
    // q/k/v/z are contiguous; after attention the whole 4x region is reused as FFN hidden
    bf16_t* qb  = (bf16_t*)alloc((size_t)kBS * kEmb * 2);
    bf16_t* kbf = (bf16_t*)alloc((size_t)kBS * kEmb * 2);
    bf16_t* vbf = (bf16_t*)alloc((size_t)kBS * kEmb * 2);
    bf16_t* zb  = (bf16_t*)alloc((size_t)kBS * kEmb * 2);
    float*  tmpF = (float*)alloc((size_t)kBS * kEmb * 4);   // attn-out, later ffn-out
    float*  h1   = (float*)alloc((size_t)kBS * kEmb * 4);
    bf16_t* h1b  = (bf16_t*)alloc((size_t)kBS * kEmb * 2);
    bf16_t* f1b  = qb;   // [kBS, kFfw] bf16 == exactly the q+k+v+z region

    auto cvt = [&](const float* src, bf16_t* dst, int n) {
        cvt_bf16_kernel<<<1024, 256, 0, stream>>>(src, dst, n);
    };
    auto gemm = [&](const bf16_t* A, const bf16_t* Bm, const float* bias,
                    float* oF, bf16_t* oB, int M, int N, int K, float alpha, int act) {
        dim3 grid(N / 128, M / 128);
        gemm_bf16_kernel<<<grid, 256, 0, stream>>>(A, Bm, bias, oF, oB, M, N, K, alpha, act);
    };

    // 1) precision conversion
    cvt(x,  xb,  kBS * kEmb);
    cvt(Wq, Wqb, kEmb * kEmb);
    cvt(Wk, Wkb, kEmb * kEmb);
    cvt(Wv, Wvb, kEmb * kEmb);
    cvt(Wo, Wob, kEmb * kEmb);
    cvt(W1, W1b, kEmb * kFfw);
    cvt(W2, W2b, kFfw * kEmb);

    // 2) QKV projections (q pre-scaled by 1/sqrt(64) = 0.125)
    gemm(xb, Wqb, bq, nullptr, qb,  kBS, kEmb, kEmb, 0.125f, 0);
    gemm(xb, Wkb, bk, nullptr, kbf, kBS, kEmb, kEmb, 1.f,    0);
    gemm(xb, Wvb, bv, nullptr, vbf, kBS, kEmb, kEmb, 1.f,    0);

    // 3) fused flash attention -> z [kBS, kEmb] bf16
    attn_kernel<<<dim3(kB * kHead * (kS / 128)), 256, 0, stream>>>(qb, kbf, vbf, mask, zb);

    // 4) output projection + residual + LN1
    gemm(zb, Wob, bo, tmpF, nullptr, kBS, kEmb, kEmb, 1.f, 0);
    add_ln_kernel<<<kBS, 256, 0, stream>>>(x, tmpF, g1, be1, h1, h1b);

    // 5) FFN: GELU(h @ W1 + b1) @ W2 + b2
    gemm(h1b, W1b, b1, nullptr, f1b, kBS, kFfw, kEmb, 1.f, 1);
    gemm(f1b, W2b, b2, tmpF, nullptr, kBS, kEmb, kFfw, 1.f, 0);

    // 6) residual + LN2 -> output (f32)
    add_ln_kernel<<<kBS, 256, 0, stream>>>(h1, tmpF, g2, be2, (float*)d_out, nullptr);
}
